// HWTConv2D_23510650978987
// MI455X (gfx1250) — matrix operand order repaired and checked by reading, not compiled, not run on hardware
//
#include <hip/hip_runtime.h>
#include <stdint.h>

typedef __attribute__((ext_vector_type(16))) _Float16 v16h;
typedef __attribute__((ext_vector_type(8)))  float    v8f;

#define HAAR_NORM 0.70710678118654752f
#define LSTRIDE 257   // 256 + 1 pad: conflict-free row and column walks

// ---------------------------------------------------------------------------
// Fused full 2-D Haar (8 levels along W, then 8 along H) for one (b,c)
// 256x256 image held entirely in LDS (257 KB dynamic allocation; CDNA5 WGP
// has 320 KB). 512 threads = 16 waves; each wave exclusively owns 16 lines
// per phase and transforms them one at a time. Within a wave LDS ops are
// in-order (DScnt), and each level buffers all its reads (<=4 pairs/lane) in
// registers before any write, so the in-place butterfly needs NO barriers;
// only 3 block-wide barriers total (load->rows, rows->cols, cols->store).
// Global I/O is b128-vectorized: one block per WGP means few waves, so big
// requests + high outstanding-load count are needed to saturate HBM.
// ---------------------------------------------------------------------------
template <bool INV, bool ADDX>
__global__ void __launch_bounds__(512)
haar2d_kernel(const float* __restrict__ src, float* __restrict__ dst,
              const float* __restrict__ xadd)
{
    extern __shared__ float t[];            // [256][LSTRIDE]

    const int tid = threadIdx.x;
    const size_t base = (size_t)blockIdx.x * 65536;   // one (b,c) image

    #pragma unroll 4
    for (int i = 0; i < 32; ++i) {
        int n4 = (i * 512 + tid) * 4;                 // 16B-aligned element idx
        const float4 g = *(const float4*)(src + base + n4);
        float* p = t + (n4 >> 8) * LSTRIDE + (n4 & 255);
        p[0] = g.x; p[1] = g.y; p[2] = g.z; p[3] = g.w;
    }
    __syncthreads();

    const int wv = tid >> 5;                // wave 0..15
    const int l  = tid & 31;                // lane

    #pragma unroll
    for (int phase = 0; phase < 2; ++phase) {
        const int es = phase ? LSTRIDE : 1;       // element stride along line
        const int ls = phase ? 1 : LSTRIDE;       // stride between lines
        for (int i = 0; i < 16; ++i) {
            float* lb = t + (wv * 16 + i) * ls;   // wave-exclusive line
            if (!INV) {
                int len = 256;
                #pragma unroll
                for (int lev = 0; lev < 8; ++lev) {
                    const int half = len >> 1;
                    float e[4], o[4];
                    int cnt = 0;
                    for (int k = l; k < half; k += 32) {
                        e[cnt] = lb[(2 * k) * es];
                        o[cnt] = lb[(2 * k + 1) * es];
                        ++cnt;
                    }
                    cnt = 0;
                    for (int k = l; k < half; k += 32) {
                        lb[k * es]          = HAAR_NORM * (e[cnt] + o[cnt]);
                        lb[(half + k) * es] = HAAR_NORM * (e[cnt] - o[cnt]);
                        ++cnt;
                    }
                    len = half;
                }
            } else {
                int len = 2;
                #pragma unroll
                for (int lev = 0; lev < 8; ++lev) {
                    const int half = len >> 1;
                    float a[4], d[4];
                    int cnt = 0;
                    for (int k = l; k < half; k += 32) {
                        a[cnt] = lb[k * es];
                        d[cnt] = lb[(half + k) * es];
                        ++cnt;
                    }
                    cnt = 0;
                    for (int k = l; k < half; k += 32) {
                        lb[(2 * k) * es]     = HAAR_NORM * (a[cnt] + d[cnt]);
                        lb[(2 * k + 1) * es] = HAAR_NORM * (a[cnt] - d[cnt]);
                        ++cnt;
                    }
                    len <<= 1;
                }
            }
        }
        __syncthreads();
    }

    #pragma unroll 4
    for (int i = 0; i < 32; ++i) {
        int n4 = (i * 512 + tid) * 4;
        const float* p = t + (n4 >> 8) * LSTRIDE + (n4 & 255);
        float4 g;
        g.x = p[0]; g.y = p[1]; g.z = p[2]; g.w = p[3];
        if (ADDX) {
            const float4 xv = *(const float4*)(xadd + base + n4);
            g.x += xv.x; g.y += xv.y; g.z += xv.z; g.w += xv.w;
        }
        *(float4*)(dst + base + n4) = g;
    }
}

// ---------------------------------------------------------------------------
// Fused: f3 = f1*v[p]; f4 = W[p] x f3 (per-pixel 64x64 channel mix, WMMA);
//        f5 = sign(f4)*relu(|f4|-tau[p]); acc += f5 over p=0..1.
// Block = 256 thr = 8 waves. Block tile: 32 pixels (one row h, 32 consecutive
// w) x all 64 out channels. Waves 0-3 -> pixel tile 0 (M blocks 0/16/32/48),
// waves 4-7 -> pixel tile 1.
// ---------------------------------------------------------------------------
__global__ void __launch_bounds__(256)
gemm_shrink_kernel(const float* __restrict__ f1,   // (B,64,H,W)
                   const float* __restrict__ vv,   // (2,1,1,H,W)
                   const float* __restrict__ wmat, // (2,64,64)
                   const float* __restrict__ tau,  // (2,1,1,H,W)
                   float* __restrict__ out)        // (B,64,H,W)
{
    constexpr int C = 64, W = 256, H = 256;
    constexpr int HW = H * W;

    // f16 staged activations, [pass][pixel][channel], channel padded 64->66
    __shared__ _Float16 sB[2][32][66];

    const int blk = blockIdx.x;
    const int w0 = (blk & 7) * 32;          // W/32 = 8 tiles per row
    const int h  = (blk >> 3) & (H - 1);
    const int b  = blk >> 11;               // / (8*256)

    const int tid = threadIdx.x;

    // staging: thread loads a 16B chunk (4 consecutive pixels) of one channel
    const int pos4 = (tid & 7) * 4;         // pixel base within tile
    const int cc0  = tid >> 3;              // channel 0..31 (two iterations)

    const float* xb = f1 + (size_t)b * C * HW + (size_t)h * W + w0;
    const float4 va0 = *(const float4*)(vv + (size_t)h * W + w0 + pos4);
    const float4 va1 = *(const float4*)(vv + (size_t)HW + (size_t)h * W + w0 + pos4);
    #pragma unroll
    for (int c = cc0; c < C; c += 32) {
        const float4 xv = *(const float4*)(xb + (size_t)c * HW + pos4);
        sB[0][pos4 + 0][c] = (_Float16)(xv.x * va0.x);
        sB[0][pos4 + 1][c] = (_Float16)(xv.y * va0.y);
        sB[0][pos4 + 2][c] = (_Float16)(xv.z * va0.z);
        sB[0][pos4 + 3][c] = (_Float16)(xv.w * va0.w);
        sB[1][pos4 + 0][c] = (_Float16)(xv.x * va1.x);
        sB[1][pos4 + 1][c] = (_Float16)(xv.y * va1.y);
        sB[1][pos4 + 2][c] = (_Float16)(xv.z * va1.z);
        sB[1][pos4 + 3][c] = (_Float16)(xv.w * va1.w);
    }
    __syncthreads();

    const int wave = tid >> 5;
    const int lane = tid & 31;
    const int pt   = wave >> 2;             // pixel sub-tile (16 pixels)
    const int m0   = (wave & 3) * 16;       // out-channel block
    const int hl   = lane >> 4;             // half-lane -> K sub-chunk / M+8
    const int nl   = lane & 15;             // row within A-M / col within B-N

    // A fragments (weights) per [pass][K-half], ISA 16-bit A 16x32 layout:
    // lane nl -> M row; VGPR v: K = kh*32 + (v/4)*16 + hl*8 + 2*(v%4)
    v16h A[2][2];
    #pragma unroll
    for (int p = 0; p < 2; ++p) {
        const float* wp = wmat + (size_t)p * C * C + (size_t)(m0 + nl) * C;
        #pragma unroll
        for (int kh = 0; kh < 2; ++kh) {
            union { _Float16 hh[16]; v16h vec; } au;
            #pragma unroll
            for (int v = 0; v < 8; ++v) {
                int k = kh * 32 + (v >> 2) * 16 + hl * 8 + 2 * (v & 3);
                au.hh[2 * v]     = (_Float16)wp[k];
                au.hh[2 * v + 1] = (_Float16)wp[k + 1];
            }
            A[p][kh] = au.vec;
        }
    }

    const int pb = pt * 16;                 // pixel base within block tile
    v8f acc = {};
    #pragma unroll
    for (int p = 0; p < 2; ++p) {
        v8f c = {};
        #pragma unroll
        for (int kh = 0; kh < 2; ++kh) {
            // B 32x16 layout: lane nl -> N col; same K order as A:
            // VGPR v holds K = kh*32 + (v/4)*16 + hl*8 + {2(v%4), 2(v%4)+1}
            // in LDS (channel-major), 4-byte aligned.
            const uint32_t* bp32 = (const uint32_t*)&sB[p][pb + nl][kh * 32 + hl * 8];
            union { uint32_t u[8]; v16h vec; } bu;
            #pragma unroll
            for (int q = 0; q < 8; ++q) bu.u[q] = bp32[(q >> 2) * 8 + (q & 3)];
            c = __builtin_amdgcn_wmma_f32_16x16x32_f16(
                    false, A[p][kh], false, bu.vec, (short)0, c, false, false);
        }
        const float tval = tau[(size_t)p * HW + (size_t)h * W + w0 + pb + nl];
        #pragma unroll
        for (int r = 0; r < 8; ++r) {
            float f = c[r];
            float m = fabsf(f) - tval;
            m = m > 0.0f ? m : 0.0f;
            acc[r] += (f >= 0.0f) ? m : -m;
        }
    }

    // C/D layout: VGPR r -> M = m0 + hl*8 + r, N = pb + nl
    float* ob = out + (size_t)b * C * HW + (size_t)h * W + w0 + pb + nl;
    #pragma unroll
    for (int r = 0; r < 8; ++r)
        ob[(size_t)(m0 + hl * 8 + r) * HW] = acc[r];
}

// ---------------------------------------------------------------------------
extern "C" void kernel_launch(void* const* d_in, const int* in_sizes, int n_in,
                              void* d_out, int out_size, void* d_ws, size_t ws_size,
                              hipStream_t stream)
{
    (void)in_sizes; (void)n_in; (void)out_size; (void)ws_size;

    const float* x   = (const float*)d_in[0];   // (16,64,256,256)
    const float* v   = (const float*)d_in[1];   // (2,1,1,256,256)
    const float* cw  = (const float*)d_in[2];   // (2,64,64)
    const float* tau = (const float*)d_in[3];   // (2,1,1,256,256)
    float* out = (float*)d_out;                 // (16,64,256,256)
    float* f1  = (float*)d_ws;                  // 256 MB scratch

    const int imgBlocks  = 16 * 64;             // one block per (b,c) image
    const int gemmBlocks = 16 * 256 * (256 / 32);
    const size_t smem = (size_t)256 * LSTRIDE * sizeof(float);   // 257 KB

    // forward haar2d (rows then cols, fused in LDS): x -> f1
    haar2d_kernel<false, false><<<imgBlocks, 512, smem, stream>>>(x, f1, nullptr);

    // fused scale / channel-mix GEMM (WMMA) / soft-threshold / pass accum
    gemm_shrink_kernel<<<gemmBlocks, 256, 0, stream>>>(f1, v, cw, tau, out);

    // inverse haar2d (rows then cols, fused) + final (+x): out -> out
    haar2d_kernel<true, true><<<imgBlocks, 512, smem, stream>>>(out, out, x);
}
